// Phi3Attention_21964462752048
// MI455X (gfx1250) — matrix operand order repaired and checked by reading, not compiled, not run on hardware
//
#include <hip/hip_runtime.h>

// ---------------------------------------------------------------------------
// Types for CDNA5 WMMA (wave32): v_wmma_f32_16x16x32_bf16
// ---------------------------------------------------------------------------
typedef __attribute__((ext_vector_type(16))) __bf16          v16bf;
typedef __attribute__((ext_vector_type(8)))  float           v8f;
typedef __attribute__((ext_vector_type(8)))  unsigned short  v8us;
typedef int v4i __attribute__((vector_size(16)));   // matches async-LDS builtin param

union FragU {
    v8us  h[2];
    v16bf b;
};

__device__ inline v16bf frag_2x8(const unsigned short* p0, const unsigned short* p1) {
    FragU u;
    u.h[0] = *(const v8us*)p0;
    u.h[1] = *(const v8us*)p1;
    return u.b;
}

__device__ inline v8f wmma_bf16(v16bf a, v16bf b, v8f c) {
    // 8 args: (neg_a, A, neg_b, B, c_mod, C, reuse_a, reuse_b)
    return __builtin_amdgcn_wmma_f32_16x16x32_bf16(false, a, false, b, (short)0, c, false, false);
}

__device__ inline v8f zero8() {
    v8f z = {0.f, 0.f, 0.f, 0.f, 0.f, 0.f, 0.f, 0.f};
    return z;
}

__device__ inline unsigned short f32_to_bf16(float f) {
    unsigned u = __builtin_bit_cast(unsigned, f);
    unsigned r = u + 0x7FFFu + ((u >> 16) & 1u);   // round-to-nearest-even
    return (unsigned short)(r >> 16);
}

__device__ inline float bf16_to_f32(unsigned short h) {
    unsigned u = ((unsigned)h) << 16;
    return __builtin_bit_cast(float, u);
}

// ---------------------------------------------------------------------------
// Async global->LDS copy (CDNA5 path, ASYNCcnt-tracked), with safe fallback.
// Builtin signature (from diagnostic): (v4i __device__* src, v4i AS3* lds,
// imm offset, imm cpol).
// ---------------------------------------------------------------------------
#if defined(__HIP_DEVICE_COMPILE__) && __has_builtin(__builtin_amdgcn_global_load_async_to_lds_b128)
#define HAVE_ASYNC_COPY 1
#else
#define HAVE_ASYNC_COPY 0
#endif

__device__ inline void async_copy_b128(const unsigned short* gsrc, unsigned short* ldst) {
#if HAVE_ASYNC_COPY
    __builtin_amdgcn_global_load_async_to_lds_b128(
        (__attribute__((address_space(1))) v4i*)gsrc,
        (__attribute__((address_space(3))) v4i*)ldst, 0, 0);
#else
    *(uint4*)ldst = *(const uint4*)gsrc;
#endif
}

__device__ inline void wait_async_copies() {
#if HAVE_ASYNC_COPY
#if __has_builtin(__builtin_amdgcn_s_wait_asynccnt)
    __builtin_amdgcn_s_wait_asynccnt(0);
#else
    asm volatile("s_wait_asynccnt 0" ::: "memory");
#endif
#endif
}

// ---------------------------------------------------------------------------
// Problem constants (match reference)
// ---------------------------------------------------------------------------
#define HIDDEN   5120
#define N_HEADS  40
#define N_KV     10
#define GQ       4          // N_HEADS / N_KV
#define HEAD_DIM 128
#define NQKV     60         // N_HEADS + 2*N_KV
#define BATCH    8
#define SEQ      1024
#define TTOK     (BATCH * SEQ)   // 8192

// ---------------------------------------------------------------------------
// f32 -> bf16 conversion (vectorized x4)
// ---------------------------------------------------------------------------
__global__ __launch_bounds__(256) void cvt_f32_bf16(const float* __restrict__ in,
                                                    unsigned short* __restrict__ out,
                                                    long long n4) {
    long long i = (long long)blockIdx.x * blockDim.x + threadIdx.x;
    if (i >= n4) return;
    long long i4 = i * 4;
    float4 v = *(const float4*)(in + i4);
    uint2 pk;
    pk.x = (unsigned)f32_to_bf16(v.x) | ((unsigned)f32_to_bf16(v.y) << 16);
    pk.y = (unsigned)f32_to_bf16(v.z) | ((unsigned)f32_to_bf16(v.w) << 16);
    *(uint2*)(out + i4) = pk;
}

// ---------------------------------------------------------------------------
// f32 -> bf16 conversion + transpose: in (K,N) row-major -> out (N,K) row-major.
// 32x32 tiles through LDS; both global sides coalesced.
// ---------------------------------------------------------------------------
__global__ __launch_bounds__(256) void cvt_transpose_f32_bf16(const float* __restrict__ in,
                                                              unsigned short* __restrict__ out,
                                                              int K, int N) {
    __shared__ unsigned short tile[32][33];
    const int k0 = blockIdx.y * 32;
    const int n0 = blockIdx.x * 32;
    const int tid = threadIdx.x;
#pragma unroll
    for (int i = 0; i < 4; ++i) {
        int e = tid + i * 256;
        int k = e >> 5, n = e & 31;
        tile[n][k] = f32_to_bf16(in[(size_t)(k0 + k) * N + n0 + n]);
    }
    __syncthreads();
#pragma unroll
    for (int i = 0; i < 4; ++i) {
        int e = tid + i * 256;
        int n = e >> 5, k = e & 31;
        out[(size_t)(n0 + n) * K + k0 + k] = tile[n][k];
    }
}

// ---------------------------------------------------------------------------
// Tiled bf16 WMMA GEMM: C(MxN) = A(MxK) * BT^T, where A is (M,K) row-major and
// BT is (N,K) row-major (i.e. B pre-transposed) -> every LDS stage is a
// contiguous async b128 copy; every WMMA fragment is two contiguous 16B
// ds_load_b128 per lane.
// Block tile 128(M) x 256(N), 256 threads = 8 waves (4x2), K-step 32.
// Wave tile 32x128: 2 A-frags, 8 B-frags, 16 WMMAs per K-step.
// ---------------------------------------------------------------------------
#define LDS_STRIDE 40   // 32 k elems + pad; 80B rows keep 16B alignment

template <int OUT_BF16>
__global__ __launch_bounds__(256) void gemm_wmma_bf16(const unsigned short* __restrict__ A,
                                                      const unsigned short* __restrict__ BT,
                                                      void* __restrict__ C,
                                                      int M, int N, int Kd) {
    __shared__ unsigned short As[128 * LDS_STRIDE];  // [row][k]
    __shared__ unsigned short Bs[256 * LDS_STRIDE];  // [col][k]

    const int tid   = threadIdx.x;
    const int lane  = tid & 31;
    const int wave  = tid >> 5;
    const int wm    = wave & 3;    // 4 row groups of 32
    const int wn    = wave >> 2;   // 2 col groups of 128
    const int l16   = lane & 15;
    const int lhalf = lane >> 4;

    const int n0 = blockIdx.x * 256;
    const int m0 = blockIdx.y * 128;

    const int akb = lhalf ? 8 : 0;    // A-frag K base within 32-chunk
    const int bkb = lhalf ? 8 : 0;    // B-frag K base within 32-chunk

    v8f acc[2][8];
#pragma unroll
    for (int sr = 0; sr < 2; ++sr)
#pragma unroll
        for (int sc = 0; sc < 8; ++sc)
            acc[sr][sc] = zero8();

    // Per-thread staging coordinates (contiguous 8-elem chunks)
    const int arow = tid >> 2;            // 0..63  (+64 on second chunk)
    const int acv  = (tid & 3) * 8;
    const int bcol = tid >> 2;            // 0..63  (+64,+128,+192)
    const int bcv  = (tid & 3) * 8;

    for (int k0 = 0; k0 < Kd; k0 += 32) {
        __syncthreads();   // previous iteration's fragment reads complete
        // Stage A tile: 128 rows x 32 k (2 async b128 per thread)
#pragma unroll
        for (int i = 0; i < 2; ++i) {
            int row = arow + i * 64;
            async_copy_b128(A + (size_t)(m0 + row) * Kd + k0 + acv,
                            &As[row * LDS_STRIDE + acv]);
        }
        // Stage B tile: 256 cols x 32 k (4 async b128 per thread)
#pragma unroll
        for (int i = 0; i < 4; ++i) {
            int col = bcol + i * 64;
            async_copy_b128(BT + (size_t)(n0 + col) * Kd + k0 + bcv,
                            &Bs[col * LDS_STRIDE + bcv]);
        }
        // Prefetch next K-step into L2 while WMMAs run
        if (k0 + 32 < Kd) {
            __builtin_prefetch(A + (size_t)(m0 + arow) * Kd + k0 + 32 + acv, 0, 1);
            __builtin_prefetch(BT + (size_t)(n0 + bcol) * Kd + k0 + 32 + bcv, 0, 1);
        }
        wait_async_copies();
        __syncthreads();

        v16bf af[2];
#pragma unroll
        for (int sr = 0; sr < 2; ++sr) {
            const unsigned short* p = &As[(wm * 32 + sr * 16 + l16) * LDS_STRIDE];
            af[sr] = frag_2x8(p + akb, p + akb + 16);
        }
#pragma unroll
        for (int sc = 0; sc < 8; ++sc) {
            const unsigned short* p = &Bs[(wn * 128 + sc * 16 + l16) * LDS_STRIDE];
            v16bf bf = frag_2x8(p + bkb, p + bkb + 16);
            acc[0][sc] = wmma_bf16(af[0], bf, acc[0][sc]);
            acc[1][sc] = wmma_bf16(af[1], bf, acc[1][sc]);
        }
    }

    // Store C. C-layout: lane -> col (N), VGPR r -> row M = r + 8*lhalf.
    for (int sr = 0; sr < 2; ++sr)
        for (int sc = 0; sc < 8; ++sc) {
            int gcol = n0 + wn * 128 + sc * 16 + l16;
#pragma unroll
            for (int r = 0; r < 8; ++r) {
                int grow = m0 + wm * 32 + sr * 16 + r + 8 * lhalf;
                if (OUT_BF16)
                    ((unsigned short*)C)[(size_t)grow * N + gcol] = f32_to_bf16(acc[sr][sc][r]);
                else
                    ((float*)C)[(size_t)grow * N + gcol] = acc[sr][sc][r];
            }
        }
}

// ---------------------------------------------------------------------------
// RoPE + split QKV (bf16 in, bf16 out). One thread per (t, n, i<64) pair.
// V is written TRANSPOSED: (B, N_KV, HEAD_DIM, SEQ) so attention can stage it
// with contiguous async copies (reduction dim = seq becomes contiguous).
// ---------------------------------------------------------------------------
__global__ __launch_bounds__(256) void rope_split(const unsigned short* __restrict__ qkv, // (T,60,128)
                                                  const int* __restrict__ pos,
                                                  unsigned short* __restrict__ Qb,  // (T,40,128)
                                                  unsigned short* __restrict__ Kb,  // (T,10,128)
                                                  unsigned short* __restrict__ VTb) // (B,10,128,SEQ)
{
    long long idx = (long long)blockIdx.x * blockDim.x + threadIdx.x;
    if (idx >= (long long)TTOK * NQKV * 64) return;
    int i = (int)(idx & 63);
    int n = (int)((idx >> 6) % NQKV);
    int t = (int)(idx / (NQKV * 64));

    const unsigned short* src = qkv + ((size_t)t * NQKV + n) * HEAD_DIM;
    unsigned short r1 = src[i];
    unsigned short r2 = src[i + 64];

    if (n < N_HEADS + N_KV) {   // rotate q and k
        float x1 = bf16_to_f32(r1);
        float x2 = bf16_to_f32(r2);
        float p    = (float)pos[t];
        float freq = __powf(10000.0f, -(float)(2 * i) * (1.0f / 128.0f));
        float ang  = p * freq;
        float s, c;
        __sincosf(ang, &s, &c);
        r1 = f32_to_bf16(x1 * c - x2 * s);
        r2 = f32_to_bf16(x2 * c + x1 * s);
    }
    if (n < N_HEADS) {
        unsigned short* dst = Qb + ((size_t)t * N_HEADS + n) * HEAD_DIM;
        dst[i] = r1; dst[i + 64] = r2;
    } else if (n < N_HEADS + N_KV) {
        unsigned short* dst = Kb + ((size_t)t * N_KV + (n - N_HEADS)) * HEAD_DIM;
        dst[i] = r1; dst[i + 64] = r2;
    } else {
        int b = t / SEQ, s = t % SEQ;
        size_t base = (size_t)(b * N_KV + (n - N_HEADS - N_KV)) * HEAD_DIM;
        VTb[(base + i) * SEQ + s]      = r1;
        VTb[(base + i + 64) * SEQ + s] = r2;
    }
}

// ---------------------------------------------------------------------------
// Flash attention (causal, GQA). One block per (batch, head, 64-row q tile).
// 128 threads = 4 waves; each wave owns 16 q rows. Key blocks of 32.
// K staged row-major (K^T B-frags contiguous per lane), V staged from the
// globally transposed layout (contiguous async copies). Online softmax with
// 16-lane shuffles; P bounced through per-wave LDS into A-layout.
// ---------------------------------------------------------------------------
#define QTILE 64
#define KS_STRIDE 136
#define VT_STRIDE 40
#define PB_STRIDE 40

__global__ __launch_bounds__(128) void attn_wmma(const unsigned short* __restrict__ Qb,
                                                 const unsigned short* __restrict__ Kb,
                                                 const unsigned short* __restrict__ VTb,
                                                 unsigned short* __restrict__ Ob) {
    const int nqt  = SEQ / QTILE;               // 16
    int blk  = blockIdx.x;
    int qt   = blk % nqt;
    int head = (blk / nqt) % N_HEADS;
    int b    = blk / (nqt * N_HEADS);
    int kv   = head / GQ;

    const int tid   = threadIdx.x;
    const int wave  = tid >> 5;
    const int lane  = tid & 31;
    const int l16   = lane & 15;
    const int lhalf = lane >> 4;

    __shared__ unsigned short Ks[32 * KS_STRIDE];     // [key][dim]
    __shared__ unsigned short Vt[128 * VT_STRIDE];    // [dim][key]
    __shared__ unsigned short Pb[4][16 * PB_STRIDE];  // per-wave P bounce

    const int akb = lhalf ? 8 : 0;
    const int bkb = lhalf ? 8 : 0;

    // Q fragments: A-layout straight from global (dims contiguous per lane)
    const int qrow0 = qt * QTILE + wave * 16;
    const unsigned short* qptr =
        Qb + ((size_t)(b * SEQ + qrow0 + l16) * N_HEADS + head) * HEAD_DIM;
    v16bf qf[4];
#pragma unroll
    for (int kc = 0; kc < 4; ++kc)
        qf[kc] = frag_2x8(qptr + kc * 32 + akb, qptr + kc * 32 + akb + 16);

    float m[8], l[8];
    v8f   o[8];
#pragma unroll
    for (int r = 0; r < 8; ++r) { m[r] = -3.0e38f; l[r] = 0.f; }
#pragma unroll
    for (int dt = 0; dt < 8; ++dt) o[dt] = zero8();

    const float scale = 0.08838834764831845f;   // 1/sqrt(128)
    const int jend = qt * QTILE + QTILE;        // uniform per block (causal bound)

    // Per-thread staging coordinates
    const int skey = tid >> 4;            // 0..7 (+8,+16,+24)
    const int scv  = (tid & 15) * 8;      // dim chunk for K
    const int vdim = tid >> 2;            // 0..31 (+32,+64,+96)
    const int vcv  = (tid & 3) * 8;       // key chunk for V

    const unsigned short* kbase_g = Kb + ((size_t)(b * SEQ) * N_KV + kv) * HEAD_DIM;
    const unsigned short* vbase_g = VTb + (size_t)(b * N_KV + kv) * HEAD_DIM * SEQ;

    for (int j0 = 0; j0 < jend; j0 += 32) {
        __syncthreads();
        // Stage K block: 32 keys x 128 dims (async, contiguous)
#pragma unroll
        for (int i = 0; i < 4; ++i) {
            int key = skey + i * 8;
            async_copy_b128(kbase_g + (size_t)(j0 + key) * N_KV * HEAD_DIM + scv,
                            &Ks[key * KS_STRIDE + scv]);
        }
        // Stage V block from transposed global: 128 dims x 32 keys (async, contiguous)
#pragma unroll
        for (int i = 0; i < 4; ++i) {
            int dim = vdim + i * 32;
            async_copy_b128(vbase_g + (size_t)dim * SEQ + j0 + vcv,
                            &Vt[dim * VT_STRIDE + vcv]);
        }
        wait_async_copies();
        __syncthreads();

        // S = Q * K^T : two 16x16 tiles (32 keys)
        float s0[8], s1[8];
#pragma unroll
        for (int nh = 0; nh < 2; ++nh) {
            v8f acc = zero8();
            const unsigned short* kp = &Ks[(nh * 16 + l16) * KS_STRIDE];
#pragma unroll
            for (int kc = 0; kc < 4; ++kc) {
                v16bf bf = frag_2x8(kp + kc * 32 + bkb, kp + kc * 32 + bkb + 16);
                acc = wmma_bf16(qf[kc], bf, acc);
            }
            float* dst = nh ? s1 : s0;
            int key = j0 + nh * 16 + l16;
#pragma unroll
            for (int r = 0; r < 8; ++r) {
                int qrow = qrow0 + r + 8 * lhalf;
                float v  = acc[r] * scale;
                dst[r]   = (key <= qrow) ? v : -3.0e38f;
            }
        }

        // Online softmax per row (rows across VGPR index; cols across 16 lanes)
#pragma unroll
        for (int r = 0; r < 8; ++r) {
            float mt = fmaxf(s0[r], s1[r]);
            for (int msk = 1; msk < 16; msk <<= 1) mt = fmaxf(mt, __shfl_xor(mt, msk, 32));
            float mn    = fmaxf(m[r], mt);
            float alpha = __expf(m[r] - mn);
            float p0    = __expf(s0[r] - mn);
            float p1    = __expf(s1[r] - mn);
            float rs    = p0 + p1;
            for (int msk = 1; msk < 16; msk <<= 1) rs += __shfl_xor(rs, msk, 32);
            l[r] = l[r] * alpha + rs;
            m[r] = mn;
#pragma unroll
            for (int dt = 0; dt < 8; ++dt) o[dt][r] *= alpha;
            s0[r] = p0;
            s1[r] = p1;
        }

        // Bounce P (C-layout) through per-wave LDS into A-layout
        unsigned short* pb = Pb[wave];
#pragma unroll
        for (int r = 0; r < 8; ++r) {
            int M = r + 8 * lhalf;
            pb[M * PB_STRIDE + l16]      = f32_to_bf16(s0[r]);
            pb[M * PB_STRIDE + 16 + l16] = f32_to_bf16(s1[r]);
        }
        asm volatile("s_wait_dscnt 0" ::: "memory");   // same-wave LDS store->load
        const unsigned short* pr = pb + l16 * PB_STRIDE;
        v16bf pf = frag_2x8(pr + akb, pr + akb + 16);

        // O += P * V  (8 dim tiles)
#pragma unroll
        for (int dt = 0; dt < 8; ++dt) {
            const unsigned short* vp = &Vt[(dt * 16 + l16) * VT_STRIDE];
            v16bf bf = frag_2x8(vp + bkb, vp + bkb + 16);
            o[dt] = wmma_bf16(pf, bf, o[dt]);
        }
    }

    // Epilogue: O /= l, store bf16 (T, N_HEADS, HEAD_DIM)
#pragma unroll
    for (int r = 0; r < 8; ++r) {
        float inv = 1.0f / l[r];
        int qrow  = qrow0 + r + 8 * lhalf;
        size_t base = ((size_t)(b * SEQ + qrow) * N_HEADS + head) * HEAD_DIM;
#pragma unroll
        for (int dt = 0; dt < 8; ++dt)
            Ob[base + dt * 16 + l16] = f32_to_bf16(o[dt][r] * inv);
    }
}

// ---------------------------------------------------------------------------
// Launch
// ---------------------------------------------------------------------------
extern "C" void kernel_launch(void* const* d_in, const int* in_sizes, int n_in,
                              void* d_out, int out_size, void* d_ws, size_t ws_size,
                              hipStream_t stream) {
    (void)in_sizes; (void)n_in; (void)out_size; (void)ws_size;

    const float* x     = (const float*)d_in[0];  // (T, HIDDEN)
    const float* w_qkv = (const float*)d_in[1];  // (HIDDEN, 60, 128)
    const float* w_o   = (const float*)d_in[2];  // (40, 128, HIDDEN)
    const int*   pos   = (const int*)d_in[3];    // (T,)
    float*       out   = (float*)d_out;          // (T, HIDDEN)

    // Workspace carve-out (bf16 buffers)
    size_t off = 0;
    auto carve = [&](size_t bytes) {
        void* p = (char*)d_ws + off;
        off += (bytes + 255) & ~(size_t)255;
        return p;
    };
    unsigned short* xb     = (unsigned short*)carve((size_t)TTOK * HIDDEN * 2);
    unsigned short* wqkvT  = (unsigned short*)carve((size_t)HIDDEN * NQKV * HEAD_DIM * 2); // (7680,5120)
    unsigned short* woT    = (unsigned short*)carve((size_t)N_HEADS * HEAD_DIM * HIDDEN * 2); // (5120,5120)
    unsigned short* qkvb   = (unsigned short*)carve((size_t)TTOK * NQKV * HEAD_DIM * 2);
    unsigned short* Qb     = (unsigned short*)carve((size_t)TTOK * N_HEADS * HEAD_DIM * 2);
    unsigned short* Kb     = (unsigned short*)carve((size_t)TTOK * N_KV * HEAD_DIM * 2);
    unsigned short* VTb    = (unsigned short*)carve((size_t)TTOK * N_KV * HEAD_DIM * 2);
    unsigned short* Ab     = (unsigned short*)carve((size_t)TTOK * N_HEADS * HEAD_DIM * 2);

    // 1. Convert x to bf16 (stays row-major: K-contiguous as A operand)
    {
        long long n4 = (long long)TTOK * HIDDEN / 4;
        cvt_f32_bf16<<<(unsigned)((n4 + 255) / 256), 256, 0, stream>>>(x, xb, n4);
    }
    // 2. Convert + transpose weights so the GEMM B operand is K-contiguous
    {
        dim3 g1((NQKV * HEAD_DIM) / 32, HIDDEN / 32);   // w_qkv (5120,7680) -> (7680,5120)
        cvt_transpose_f32_bf16<<<g1, 256, 0, stream>>>(w_qkv, wqkvT, HIDDEN, NQKV * HEAD_DIM);
        dim3 g2(HIDDEN / 32, (N_HEADS * HEAD_DIM) / 32); // w_o (5120,5120) -> (5120,5120)^T
        cvt_transpose_f32_bf16<<<g2, 256, 0, stream>>>(w_o, woT, N_HEADS * HEAD_DIM, HIDDEN);
    }

    // 3. QKV projection: (T x 5120) * (5120 x 7680) -> qkv bf16
    {
        dim3 grid((NQKV * HEAD_DIM) / 256, TTOK / 128);
        gemm_wmma_bf16<1><<<grid, 256, 0, stream>>>(xb, wqkvT, qkvb,
                                                    TTOK, NQKV * HEAD_DIM, HIDDEN);
    }

    // 4. RoPE + split (V written transposed per (b, kv): (HEAD_DIM, SEQ))
    {
        long long n = (long long)TTOK * NQKV * 64;
        rope_split<<<(unsigned)((n + 255) / 256), 256, 0, stream>>>(qkvb, pos, Qb, Kb, VTb);
    }

    // 5. Flash attention
    {
        unsigned blocks = (unsigned)(BATCH * N_HEADS * (SEQ / QTILE));
        attn_wmma<<<blocks, 128, 0, stream>>>(Qb, Kb, VTb, Ab);
    }

    // 6. Output projection: (T x 5120) * (5120 x 5120) -> f32 out
    {
        dim3 grid(HIDDEN / 256, TTOK / 128);
        gemm_wmma_bf16<0><<<grid, 256, 0, stream>>>(Ab, woT, out,
                                                    TTOK, HIDDEN, N_HEADS * HEAD_DIM);
    }
}
